// UltraGNNBackbone_28595892257411
// MI455X (gfx1250) — matrix operand order repaired and checked by reading, not compiled, not run on hardware
//
#include <hip/hip_runtime.h>
#include <hip/hip_bf16.h>
#include <math.h>

// ============================================================================
// UltraGNN backbone forward for MI455X (gfx1250), wave32 + WMMA.
//
// Input flattening assumption (JAX pytree order = dict keys sorted):
//   [0]   edge_index  (int32 [2,E]; row0 = src, row1 = dst)
//   [1-4] fus_W, fus_b, fus_be, fus_g
//   [5-8] in_W, in_b, in_be, in_g
//   [9 + 20*L + k] layer L param k, k in alphabetical order:
//       0 att_dst  1 att_src  2 e_W1  3 e_W2  4 e_b1  5 e_b2
//       6 f_W1     7 f_W2     8 f_b1  9 f_b2 10 gat_W 11 gat_b
//      12 gcn_W   13 gcn_b  14 post_b 15 post_g 16 pre_b 17 pre_g
//      18 se_W1   19 se_W2
//   [129..132] proj_W[0..3]   [133..136] proj_b[0..3]   [137] x
// ============================================================================

typedef __attribute__((ext_vector_type(16))) _Float16 v16h;
typedef __attribute__((ext_vector_type(8)))  float    v8f;

#define NODES 203
#define HID   320
#define HEADS 8
#define HD    40
#define NLAYERS 6

__device__ __forceinline__ float gelu_f(float x) {
  return 0.5f * x * (1.0f + erff(x * 0.7071067811865476f));
}
__device__ __forceinline__ float sigm_f(float x) {
  return 1.0f / (1.0f + expf(-x));
}
// order-preserving float<->uint for atomicMax on floats
__device__ __forceinline__ unsigned ford(float x) {
  unsigned b = __float_as_uint(x);
  return (b & 0x80000000u) ? ~b : (b | 0x80000000u);
}
__device__ __forceinline__ float orddec(unsigned u) {
  unsigned b = (u & 0x80000000u) ? (u ^ 0x80000000u) : ~u;
  return __uint_as_float(b);
}

// ---------------------------------------------------------------------------
// Generic WMMA GEMM: C[M,N] = act(A[M,K] @ B[K,N] + bias)
// A,B f16 row-major; f32 accumulate; out f32 or f16. Requires M%128==0,
// N%64==0, K%64==0 (true for every call site: K in {320,1280,960,64? no},
// all K here are 320/960/1280 -> %64==0).
// Block = 256 thr = 8 waves; block tile 128x64; BK=64; wave tile 32x32.
// B tile stored TRANSPOSED in LDS (Bt[n][k], padded) so both A and B
// fragments are contiguous 16-byte ds_load_b128 reads.
// ---------------------------------------------------------------------------
#define APAD 72  // 64 + 8 halves: row stride 144B -> conflict-free frag reads
__global__ __launch_bounds__(256) void gemm_wmma(
    const _Float16* __restrict__ A, int lda,
    const _Float16* __restrict__ B, int ldb,
    const float* __restrict__ bias,
    float* __restrict__ Cf, _Float16* __restrict__ Ch,
    int M, int N, int K, int act, int out16) {
  __shared__ _Float16 As[128][APAD];  // 18 KB
  __shared__ _Float16 Bt[64][APAD];   //  9 KB (transposed B tile)
  const int t = threadIdx.x;
  const int bm = blockIdx.y * 128;
  const int bn = blockIdx.x * 64;
  const int w = t >> 5, l = t & 31;
  const int wm = (w >> 1) * 32;  // 4 row groups of 32
  const int wn = (w & 1) * 32;   // 2 col groups of 32
  v8f acc[2][2];
  v8f zero = {};
  for (int i = 0; i < 2; ++i)
    for (int j = 0; j < 2; ++j) acc[i][j] = zero;

  const int arow = t >> 1, acol = (t & 1) * 32;   // 128x64 tile, 32 halves/thr
  const int brow = t >> 2, bcol = (t & 3) * 16;   // 64x64 tile,  16 halves/thr
  const int m16 = l & 15, hi = l >> 4;
  const int kg = hi * 8, kb = hi * 16;

  for (int k0 = 0; k0 < K; k0 += 64) {
    // ---- load A panel (128x64), 4x 128-bit per thread ----
    const _Float16* ap = A + (size_t)(bm + arow) * lda + (k0 + acol);
    const uint4* apv = (const uint4*)ap;
    uint4* asv = (uint4*)&As[arow][acol];
#pragma unroll
    for (int j = 0; j < 4; ++j) asv[j] = apv[j];
    if (k0 + 64 < K) __builtin_prefetch(ap + lda, 0, 0);  // -> global_prefetch
    // ---- load B panel (64x64) transposed into LDS ----
    const _Float16* bp = B + (size_t)(k0 + brow) * ldb + (bn + bcol);
    union { uint4 v[2]; _Float16 h[16]; } bu;
    bu.v[0] = ((const uint4*)bp)[0];
    bu.v[1] = ((const uint4*)bp)[1];
#pragma unroll
    for (int j = 0; j < 16; ++j) Bt[bcol + j][brow] = bu.h[j];
    __syncthreads();
#pragma unroll
    for (int kk = 0; kk < 64; kk += 32) {
#pragma unroll
      for (int mt = 0; mt < 2; ++mt) {
        v16h a;
        const _Float16* arp = &As[wm + mt * 16 + m16][kk + kg];
#pragma unroll
        for (int j = 0; j < 8; ++j) { a[j] = arp[j]; a[8 + j] = arp[16 + j]; }
#pragma unroll
        for (int nt = 0; nt < 2; ++nt) {
          v16h b;
          const _Float16* brp = &Bt[wn + nt * 16 + m16][kk + kg];
#pragma unroll
          for (int j = 0; j < 8; ++j) { b[j] = brp[j]; b[8 + j] = brp[16 + j]; }
          acc[mt][nt] = __builtin_amdgcn_wmma_f32_16x16x32_f16(
              false, a, false, b, (short)0, acc[mt][nt], false, false);
        }
      }
    }
    __syncthreads();
  }
  // epilogue: C layout — lane gives col (l&15), vgpr r gives row r + 8*(l>>4)
#pragma unroll
  for (int mt = 0; mt < 2; ++mt)
#pragma unroll
    for (int nt = 0; nt < 2; ++nt) {
      const int col = bn + wn + nt * 16 + m16;
      const float bv = bias ? bias[col] : 0.0f;
#pragma unroll
      for (int r = 0; r < 8; ++r) {
        const int row = bm + wm + mt * 16 + hi * 8 + r;
        float v = acc[mt][nt][r] + bv;
        if (act) v = gelu_f(v);
        if (out16) Ch[(size_t)row * N + col] = (_Float16)v;
        else       Cf[(size_t)row * N + col] = v;
      }
    }
}

// ---------------------------------------------------------------------------
// Edge-weight MLP: ew[e] = sigmoid(GELU([hn[src]||hn[dst]] @ W1 + b1) @ W2 + b2)
// Block handles 64 edges; gathers K-chunks of the virtual [E,640] matrix into
// LDS (f32->f16) and runs WMMA against W1 (640x64, f16; tile transposed in
// LDS). Second layer (64->1) is a per-edge VALU reduction. E % 64 == 0.
// ---------------------------------------------------------------------------
#define EPAD 40  // 32 + 8 halves: row stride 80B
__global__ __launch_bounds__(256) void edge_mlp_wmma(
    const float* __restrict__ hn, const int* __restrict__ src,
    const int* __restrict__ dst, const _Float16* __restrict__ W1,
    const float* __restrict__ b1, const float* __restrict__ W2,
    const float* __restrict__ b2, float* __restrict__ ew) {
  __shared__ _Float16 As[64][EPAD];
  __shared__ _Float16 Bt[64][EPAD];  // transposed W1 chunk: Bt[n][k]
  __shared__ float Hid[64][65];
  const int t = threadIdx.x;
  const int e0 = blockIdx.x * 64;
  const int w = t >> 5, l = t & 31;
  const int wr = (w >> 1) * 16;  // 4 row tiles of 16 edges
  const int wc = (w & 1) * 32;   // 2 col halves of 32
  v8f acc[2];
  v8f zero = {};
  acc[0] = zero; acc[1] = zero;

  const int arow = t >> 2, acol = (t & 3) * 8;  // 64x32 gather, 8/thr
  const int brow = t >> 3, bcol = (t & 7) * 8;  // 32x64, 8/thr
  const int m16 = l & 15, hi = l >> 4;
  const int kg = hi * 8, kb = hi * 16;

  for (int kc = 0; kc < 2 * HID; kc += 32) {
    // gather: k<320 -> hn[src], else hn[dst] (chunk never straddles 320)
    const int e = e0 + arow;
    const int k = kc + acol;
    const int node = (k < HID) ? src[e] : dst[e];
    const int ko = (k < HID) ? k : (k - HID);
    const float* hp = hn + (size_t)node * HID + ko;
#pragma unroll
    for (int j = 0; j < 8; ++j) As[arow][acol + j] = (_Float16)hp[j];
    const _Float16* bp = W1 + (size_t)(kc + brow) * 64 + bcol;
    union { uint4 v; _Float16 h[8]; } bu;
    bu.v = *(const uint4*)bp;
#pragma unroll
    for (int j = 0; j < 8; ++j) Bt[bcol + j][brow] = bu.h[j];
    __syncthreads();
    v16h a;
    const _Float16* arp = &As[wr + m16][kg];
#pragma unroll
    for (int j = 0; j < 8; ++j) { a[j] = arp[j]; a[8 + j] = arp[16 + j]; }
#pragma unroll
    for (int nt = 0; nt < 2; ++nt) {
      v16h b;
      const _Float16* brp = &Bt[wc + nt * 16 + m16][kg];
#pragma unroll
      for (int j = 0; j < 8; ++j) { b[j] = brp[j]; b[8 + j] = brp[16 + j]; }
      acc[nt] = __builtin_amdgcn_wmma_f32_16x16x32_f16(
          false, a, false, b, (short)0, acc[nt], false, false);
    }
    __syncthreads();
  }
#pragma unroll
  for (int nt = 0; nt < 2; ++nt) {
    const int col = wc + nt * 16 + m16;
    const float bv = b1[col];
#pragma unroll
    for (int r = 0; r < 8; ++r) {
      const int row = wr + hi * 8 + r;
      Hid[row][col] = gelu_f(acc[nt][r] + bv);
    }
  }
  __syncthreads();
  if (t < 64) {
    float s = 0.0f;
#pragma unroll
    for (int j = 0; j < 64; ++j) s += Hid[t][j] * W2[j];
    ew[e0 + t] = sigm_f(s + b2[0]);
  }
}

// ---------------------------------------------------------------------------
// LayerNorm over 320 channels; one wave32 per row (10 elems/lane).
// Optional GELU; writes f32 and/or f16 outputs.
// ---------------------------------------------------------------------------
__global__ __launch_bounds__(256) void ln_kernel(
    const float* __restrict__ x, const float* __restrict__ g,
    const float* __restrict__ b, float* __restrict__ o32,
    _Float16* __restrict__ o16, int rows, int act) {
  const int w = threadIdx.x >> 5, l = threadIdx.x & 31;
  const int row = blockIdx.x * 8 + w;
  if (row >= rows) return;
  const float* xr = x + (size_t)row * HID;
  float vals[10];
  float s = 0.0f;
#pragma unroll
  for (int i = 0; i < 10; ++i) { vals[i] = xr[l + i * 32]; s += vals[i]; }
#pragma unroll
  for (int off = 16; off > 0; off >>= 1) s += __shfl_xor(s, off, 32);
  const float m = s / (float)HID;
  float v = 0.0f;
#pragma unroll
  for (int i = 0; i < 10; ++i) { float d = vals[i] - m; v += d * d; }
#pragma unroll
  for (int off = 16; off > 0; off >>= 1) v += __shfl_xor(v, off, 32);
  const float inv = rsqrtf(v / (float)HID + 1e-5f);
#pragma unroll
  for (int i = 0; i < 10; ++i) {
    const int c = l + i * 32;
    float y = (vals[i] - m) * inv * g[c] + b[c];
    if (act) y = gelu_f(y);
    if (o32) o32[(size_t)row * HID + c] = y;
    if (o16) o16[(size_t)row * HID + c] = (_Float16)y;
  }
}

// ----------------------------- small helpers -------------------------------
__global__ void conv_f32_f16(const float* __restrict__ s, _Float16* __restrict__ d, size_t n) {
  size_t i = (size_t)blockIdx.x * blockDim.x + threadIdx.x;
  if (i < n) d[i] = (_Float16)s[i];
}
__global__ void fill_f32(float* p, float v, size_t n) {
  size_t i = (size_t)blockIdx.x * blockDim.x + threadIdx.x;
  if (i < n) p[i] = v;
}
__global__ void fill_u32(unsigned* p, unsigned v, size_t n) {
  size_t i = (size_t)blockIdx.x * blockDim.x + threadIdx.x;
  if (i < n) p[i] = v;
}

// ------------------------------- GCN pieces --------------------------------
__global__ void deg_scatter(const int* __restrict__ dst, const float* __restrict__ ew,
                            float* __restrict__ deg, int E) {
  int e = blockIdx.x * blockDim.x + threadIdx.x;
  if (e < E) atomicAdd(&deg[dst[e]], ew[e]);
}
__global__ void deg_to_dis(float* deg, int N) {
  int i = blockIdx.x * blockDim.x + threadIdx.x;
  if (i < N) { float d = deg[i]; deg[i] = (d > 0.0f) ? rsqrtf(d) : 0.0f; }
}
__global__ void norm_kernel(const int* __restrict__ src, const int* __restrict__ dst,
                            const float* __restrict__ ew, const float* __restrict__ dis,
                            float* __restrict__ nrm, int E) {
  int e = blockIdx.x * blockDim.x + threadIdx.x;
  if (e < E) nrm[e] = dis[src[e]] * ew[e] * dis[dst[e]];
}
// xdyn[i][c] = gcn_b[c] + dis[i]^2 * hW[i][c]  (self-loop folded in)
__global__ void xdyn_init(const float* __restrict__ hW, const float* __restrict__ dis,
                          const float* __restrict__ gcn_b, float* __restrict__ xd) {
  int i = blockIdx.x, c = threadIdx.x;
  float d = dis[i];
  size_t idx = (size_t)i * HID + c;
  xd[idx] = gcn_b[c] + d * d * hW[idx];
}
__global__ void gcn_scatter(const int* __restrict__ src, const int* __restrict__ dst,
                            const float* __restrict__ nrm, const float* __restrict__ hW,
                            float* __restrict__ xd) {
  int e = blockIdx.x, c = threadIdx.x;
  int s = src[e], d = dst[e];
  atomicAdd(&xd[(size_t)d * HID + c], nrm[e] * hW[(size_t)s * HID + c]);
}

// ------------------------------- GAT pieces --------------------------------
__global__ void attn_dots(const float* __restrict__ g, const float* __restrict__ a_src,
                          const float* __restrict__ a_dst, float* __restrict__ asrc,
                          float* __restrict__ adst, int N) {
  int idx = blockIdx.x * blockDim.x + threadIdx.x;
  if (idx >= N * HEADS) return;
  int i = idx >> 3, h = idx & 7;
  const float* gp = g + (size_t)i * HID + h * HD;
  const float* ws = a_src + h * HD;
  const float* wd = a_dst + h * HD;
  float s = 0.0f, d = 0.0f;
#pragma unroll
  for (int k = 0; k < HD; ++k) { s += gp[k] * ws[k]; d += gp[k] * wd[k]; }
  asrc[idx] = s; adst[idx] = d;
}
__global__ void gat_edge_max(const int* __restrict__ src, const int* __restrict__ dst,
                             const float* __restrict__ asrc, const float* __restrict__ adst,
                             float* __restrict__ ev, unsigned* __restrict__ mEnc,
                             int E, int N) {
  int idx = blockIdx.x * blockDim.x + threadIdx.x;
  if (idx >= (E + N) * HEADS) return;
  int entry = idx >> 3, h = idx & 7;
  int s, d;
  if (entry < E) { s = src[entry]; d = dst[entry]; } else { s = d = entry - E; }
  float e = asrc[s * HEADS + h] + adst[d * HEADS + h];
  e = (e > 0.0f) ? e : 0.2f * e;  // leaky_relu 0.2
  ev[idx] = e;
  atomicMax(&mEnc[d * HEADS + h], ford(e));
}
__global__ void gat_exp_sum(const int* __restrict__ dst, float* __restrict__ ev,
                            const unsigned* __restrict__ mEnc, float* __restrict__ ssum,
                            int E, int N) {
  int idx = blockIdx.x * blockDim.x + threadIdx.x;
  if (idx >= (E + N) * HEADS) return;
  int entry = idx >> 3, h = idx & 7;
  int d = (entry < E) ? dst[entry] : (entry - E);
  float ee = expf(ev[idx] - orddec(mEnc[d * HEADS + h]));
  ev[idx] = ee;
  atomicAdd(&ssum[d * HEADS + h], ee);
}
__global__ void xgat_init(const float* __restrict__ gat_b, float* __restrict__ xg) {
  int i = blockIdx.x, c = threadIdx.x;
  xg[(size_t)i * HID + c] = gat_b[c];
}
__global__ void gat_scatter(const int* __restrict__ src, const int* __restrict__ dst,
                            const float* __restrict__ ev, const float* __restrict__ ssum,
                            const float* __restrict__ g, float* __restrict__ xg,
                            int E, int N) {
  int idx = blockIdx.x * blockDim.x + threadIdx.x;
  if (idx >= (E + N) * HEADS) return;
  int entry = idx >> 3, h = idx & 7;
  int s, d;
  if (entry < E) { s = src[entry]; d = dst[entry]; } else { s = d = entry - E; }
  float alpha = ev[idx] / ssum[d * HEADS + h];
  const float* gp = g + (size_t)s * HID + h * HD;
  float* xp = xg + (size_t)d * HID + h * HD;
#pragma unroll 8
  for (int k = 0; k < HD; ++k) atomicAdd(&xp[k], alpha * gp[k]);
}

// ------------------------------ elementwise --------------------------------
__global__ void add3_kernel(float* __restrict__ h, const float* __restrict__ a,
                            const float* __restrict__ b) {
  size_t idx = (size_t)blockIdx.x * HID + threadIdx.x;
  h[idx] = h[idx] + a[idx] + b[idx];
}
__global__ void add2_kernel(float* __restrict__ h, const float* __restrict__ a) {
  size_t idx = (size_t)blockIdx.x * HID + threadIdx.x;
  h[idx] += a[idx];
}

// -------------------------------- SE layer ---------------------------------
__global__ void se_reduce(const float* __restrict__ h, float* __restrict__ s, int B) {
  int idx = blockIdx.x * blockDim.x + threadIdx.x;
  if (idx >= B * HID) return;
  int b = idx / HID, c = idx % HID;
  float acc = 0.0f;
  for (int n = 0; n < NODES; ++n) acc += h[((size_t)(b * NODES + n)) * HID + c];
  s[idx] = acc / (float)NODES;
}
__global__ void se_mlp1(const float* __restrict__ s, const float* __restrict__ W1,
                        float* __restrict__ sh, int B) {
  int idx = blockIdx.x * blockDim.x + threadIdx.x;
  if (idx >= B * 20) return;
  int b = idx / 20, j = idx % 20;
  float acc = 0.0f;
  const float* sp = s + (size_t)b * HID;
  for (int c = 0; c < HID; ++c) acc += sp[c] * W1[(size_t)c * 20 + j];
  sh[idx] = gelu_f(acc);
}
__global__ void se_mlp2(const float* __restrict__ sh, const float* __restrict__ W2,
                        float* __restrict__ y, int B) {
  int idx = blockIdx.x * blockDim.x + threadIdx.x;
  if (idx >= B * HID) return;
  int b = idx / HID, c = idx % HID;
  float acc = 0.0f;
  const float* sp = sh + (size_t)b * 20;
#pragma unroll
  for (int j = 0; j < 20; ++j) acc += sp[j] * W2[(size_t)j * HID + c];
  y[idx] = sigm_f(acc);
}
__global__ void se_scale(float* __restrict__ h, const float* __restrict__ y) {
  int i = blockIdx.x, c = threadIdx.x;
  int b = i / NODES;
  h[(size_t)i * HID + c] *= y[(size_t)b * HID + c];
}

// ----------------------- pyramid pooling / fusion --------------------------
__global__ void poolw_kernel(float* __restrict__ pw) {  // [4][203]
  int idx = blockIdx.x * blockDim.x + threadIdx.x;
  if (idx >= 4 * NODES) return;
  int lev = idx / NODES, pos = idx % NODES;
  const int outs[4] = {203, 101, 50, 25};  // NODES // {1,2,4,8}
  int o = outs[lev];
  float acc = 0.0f;
  for (int i = 0; i < o; ++i) {
    int s = (i * NODES) / o;
    int e = ((i + 1) * NODES + o - 1) / o;  // ceil
    if (pos >= s && pos < e) acc += 1.0f / (float)(e - s);
  }
  pw[idx] = acc / (float)o;
}
__global__ void pooled_kernel(const float* __restrict__ h, const float* __restrict__ pw,
                              float* __restrict__ pooled, int B) {
  int idx = blockIdx.x * blockDim.x + threadIdx.x;
  if (idx >= 4 * B * HID) return;
  int lev = idx / (B * HID);
  int rem = idx % (B * HID);
  int b = rem / HID, c = rem % HID;
  const float* p = pw + lev * NODES;
  float s = 0.0f;
  for (int n = 0; n < NODES; ++n) s += p[n] * h[((size_t)(b * NODES + n)) * HID + c];
  pooled[idx] = s;
}
__global__ void proj_kernel(const float* __restrict__ pooled,
                            const float* __restrict__ W0, const float* __restrict__ W1,
                            const float* __restrict__ W2, const float* __restrict__ W3,
                            const float* __restrict__ b0, const float* __restrict__ b1,
                            const float* __restrict__ b2, const float* __restrict__ b3,
                            float* __restrict__ fused, int B) {
  int idx = blockIdx.x * blockDim.x + threadIdx.x;
  if (idx >= B * 4 * 80) return;
  int b = idx / 320;
  int rem = idx % 320;
  int lev = rem / 80, oo = rem % 80;
  const float* W = (lev == 0) ? W0 : (lev == 1) ? W1 : (lev == 2) ? W2 : W3;
  const float* bb = (lev == 0) ? b0 : (lev == 1) ? b1 : (lev == 2) ? b2 : b3;
  const float* p = pooled + ((size_t)lev * B + b) * HID;
  float s = bb[oo];
  for (int c = 0; c < HID; ++c) s += p[c] * W[(size_t)c * 80 + oo];
  fused[(size_t)b * 960 + lev * 80 + oo] = s;
}
__global__ void gmeanmax_kernel(const float* __restrict__ h, float* __restrict__ fused, int B) {
  int idx = blockIdx.x * blockDim.x + threadIdx.x;
  if (idx >= B * HID) return;
  int b = idx / HID, c = idx % HID;
  float s = 0.0f, mx = -3.402823466e+38f;
  for (int n = 0; n < NODES; ++n) {
    float v = h[((size_t)(b * NODES + n)) * HID + c];
    s += v; mx = fmaxf(mx, v);
  }
  fused[(size_t)b * 960 + 320 + c] = s / (float)NODES;
  fused[(size_t)b * 960 + 640 + c] = mx;
}

// ============================================================================
// Host orchestration
// ============================================================================
static inline int cdiv(long long a, long long b) { return (int)((a + b - 1) / b); }

extern "C" void kernel_launch(void* const* d_in, const int* in_sizes, int n_in,
                              void* d_out, int out_size, void* d_ws, size_t ws_size,
                              hipStream_t stream) {
  (void)out_size; (void)ws_size;
  // ---- input mapping (see header comment) ----
  const int ix_x = n_in - 1;               // 'x' is last in sorted pytree
  const int E = in_sizes[0] / 2;
  const int Ntot = in_sizes[ix_x] / 1280;  // 25984
  const int B = Ntot / NODES;              // 128

  const int* src = (const int*)d_in[0];
  const int* dst = src + E;
  const float* fus_W = (const float*)d_in[1];
  const float* fus_b = (const float*)d_in[2];
  const float* fus_be = (const float*)d_in[3];
  const float* fus_g = (const float*)d_in[4];
  const float* in_W = (const float*)d_in[5];
  const float* in_b = (const float*)d_in[6];
  const float* in_be = (const float*)d_in[7];
  const float* in_g = (const float*)d_in[8];
  const float* projW[4], *projB[4];
  for (int i = 0; i < 4; ++i) { projW[i] = (const float*)d_in[129 + i]; projB[i] = (const float*)d_in[133 + i]; }
  const float* x_in = (const float*)d_in[ix_x];
  auto LP = [&](int L, int k) -> const float* { return (const float*)d_in[9 + 20 * L + k]; };
  // layer param slots
  enum { P_ATT_DST=0,P_ATT_SRC,P_EW1,P_EW2,P_EB1,P_EB2,P_FW1,P_FW2,P_FB1,P_FB2,
         P_GATW,P_GATB,P_GCNW,P_GCNB,P_POSTB,P_POSTG,P_PREB,P_PREG,P_SEW1,P_SEW2 };

  // ---- workspace carve-up ----
  char* ws = (char*)d_ws;
  size_t off = 0;
  auto carve = [&](size_t bytes) -> char* {
    char* p = ws + off;
    off = (off + bytes + 255) & ~(size_t)255;
    return p;
  };
  const size_t NH = (size_t)Ntot * HID;
  float*    b_h     = (float*)carve(NH * 4);
  float*    b_hn    = (float*)carve(NH * 4);
  _Float16* b_hn16  = (_Float16*)carve(NH * 2);
  float*    b_t0    = (float*)carve(NH * 4);          // hW / ffn-out
  float*    b_t1    = (float*)carve(NH * 4);          // g (gat features)
  float*    b_xdyn  = (float*)carve(NH * 4);
  float*    b_xgat  = (float*)carve(NH * 4);
  _Float16* b_big16 = (_Float16*)carve((size_t)Ntot * 1280 * 2);  // x16 then ffn hidden
  _Float16* b_w16   = (_Float16*)carve((size_t)320 * 1280 * 2);   // current f16 weight
  float*    b_ew    = (float*)carve((size_t)E * 4);
  float*    b_nrm   = (float*)carve((size_t)E * 4);
  float*    b_deg   = (float*)carve((size_t)Ntot * 4);            // becomes dis
  float*    b_ev    = (float*)carve((size_t)(E + Ntot) * HEADS * 4);
  unsigned* b_m     = (unsigned*)carve((size_t)Ntot * HEADS * 4);
  float*    b_ssum  = (float*)carve((size_t)Ntot * HEADS * 4);
  float*    b_asrc  = (float*)carve((size_t)Ntot * HEADS * 4);
  float*    b_adst  = (float*)carve((size_t)Ntot * HEADS * 4);
  float*    b_s     = (float*)carve((size_t)B * HID * 4);
  float*    b_seh   = (float*)carve((size_t)B * 20 * 4);
  float*    b_y     = (float*)carve((size_t)B * HID * 4);
  float*    b_pw    = (float*)carve(4 * NODES * 4);
  float*    b_pool  = (float*)carve((size_t)4 * B * HID * 4);
  float*    b_fused = (float*)carve((size_t)B * 960 * 4);
  _Float16* b_fus16 = (_Float16*)carve((size_t)B * 960 * 2);
  float*    b_fo    = (float*)carve((size_t)B * HID * 4);

  const int T = 256;
  auto conv = [&](const float* s, _Float16* d, size_t n) {
    conv_f32_f16<<<cdiv(n, T), T, 0, stream>>>(s, d, n);
  };
  auto gemm = [&](const _Float16* A, const _Float16* Bm, const float* bias,
                  float* Cf, _Float16* Ch, int M, int N, int K, int act, int out16) {
    gemm_wmma<<<dim3(N / 64, M / 128), 256, 0, stream>>>(A, K, Bm, N, bias, Cf, Ch,
                                                         M, N, K, act, out16);
  };

  // ================= input projection =================
  conv(x_in, b_big16, (size_t)Ntot * 1280);
  conv(in_W, b_w16, (size_t)1280 * HID);
  gemm(b_big16, b_w16, in_b, b_t0, nullptr, Ntot, HID, 1280, 0, 0);
  ln_kernel<<<cdiv(Ntot, 8), 256, 0, stream>>>(b_t0, in_g, in_be, b_h, nullptr, Ntot, 1);

  // ================= layers =================
  for (int L = 0; L < NLAYERS; ++L) {
    // pre-LN -> hn (f32 for gathers) + hn16 (f16 for GEMMs)
    ln_kernel<<<cdiv(Ntot, 8), 256, 0, stream>>>(b_h, LP(L, P_PREG), LP(L, P_PREB),
                                                 b_hn, b_hn16, Ntot, 0);
    // hW = hn @ gcn_W ; g = hn @ gat_W
    conv(LP(L, P_GCNW), b_w16, (size_t)HID * HID);
    gemm(b_hn16, b_w16, nullptr, b_t0, nullptr, Ntot, HID, HID, 0, 0);
    conv(LP(L, P_GATW), b_w16, (size_t)HID * HID);
    gemm(b_hn16, b_w16, nullptr, b_t1, nullptr, Ntot, HID, HID, 0, 0);
    // edge weights
    conv(LP(L, P_EW1), b_w16, (size_t)2 * HID * 64);
    edge_mlp_wmma<<<E / 64, 256, 0, stream>>>(b_hn, src, dst, b_w16, LP(L, P_EB1),
                                              LP(L, P_EW2), LP(L, P_EB2), b_ew);
    // degrees (init 1.0 covers the self-loop weight), dis = deg^-1/2
    fill_f32<<<cdiv(Ntot, T), T, 0, stream>>>(b_deg, 1.0f, Ntot);
    deg_scatter<<<cdiv(E, T), T, 0, stream>>>(dst, b_ew, b_deg, E);
    deg_to_dis<<<cdiv(Ntot, T), T, 0, stream>>>(b_deg, Ntot);
    norm_kernel<<<cdiv(E, T), T, 0, stream>>>(src, dst, b_ew, b_deg, b_nrm, E);
    // x_dyn = gcn_b + dis^2*hW (self loop) + scatter over edges
    xdyn_init<<<Ntot, HID, 0, stream>>>(b_t0, b_deg, LP(L, P_GCNB), b_xdyn);
    gcn_scatter<<<E, HID, 0, stream>>>(src, dst, b_nrm, b_t0, b_xdyn);
    // GAT attention
    attn_dots<<<cdiv((size_t)Ntot * HEADS, T), T, 0, stream>>>(
        b_t1, LP(L, P_ATT_SRC), LP(L, P_ATT_DST), b_asrc, b_adst, Ntot);
    fill_u32<<<cdiv((size_t)Ntot * HEADS, T), T, 0, stream>>>(b_m, 0u, (size_t)Ntot * HEADS);
    gat_edge_max<<<cdiv((size_t)(E + Ntot) * HEADS, T), T, 0, stream>>>(
        src, dst, b_asrc, b_adst, b_ev, b_m, E, Ntot);
    fill_f32<<<cdiv((size_t)Ntot * HEADS, T), T, 0, stream>>>(b_ssum, 0.0f, (size_t)Ntot * HEADS);
    gat_exp_sum<<<cdiv((size_t)(E + Ntot) * HEADS, T), T, 0, stream>>>(
        dst, b_ev, b_m, b_ssum, E, Ntot);
    xgat_init<<<Ntot, HID, 0, stream>>>(LP(L, P_GATB), b_xgat);
    gat_scatter<<<cdiv((size_t)(E + Ntot) * HEADS, T), T, 0, stream>>>(
        src, dst, b_ev, b_ssum, b_t1, b_xgat, E, Ntot);
    // h = x_dyn + x_gat + identity
    add3_kernel<<<Ntot, HID, 0, stream>>>(b_h, b_xdyn, b_xgat);
    // FFN: post-LN -> f16; GEMM1 (GELU, f16 out) -> GEMM2; residual
    ln_kernel<<<cdiv(Ntot, 8), 256, 0, stream>>>(b_h, LP(L, P_POSTG), LP(L, P_POSTB),
                                                 nullptr, b_hn16, Ntot, 0);
    conv(LP(L, P_FW1), b_w16, (size_t)HID * 1280);
    gemm(b_hn16, b_w16, LP(L, P_FB1), nullptr, b_big16, Ntot, 1280, HID, 1, 1);
    conv(LP(L, P_FW2), b_w16, (size_t)1280 * HID);
    gemm(b_big16, b_w16, LP(L, P_FB2), b_t0, nullptr, Ntot, HID, 1280, 0, 0);
    add2_kernel<<<Ntot, HID, 0, stream>>>(b_h, b_t0);
    // SE
    se_reduce<<<cdiv((size_t)B * HID, T), T, 0, stream>>>(b_h, b_s, B);
    se_mlp1<<<cdiv((size_t)B * 20, T), T, 0, stream>>>(b_s, LP(L, P_SEW1), b_seh, B);
    se_mlp2<<<cdiv((size_t)B * HID, T), T, 0, stream>>>(b_seh, LP(L, P_SEW2), b_y, B);
    se_scale<<<Ntot, HID, 0, stream>>>(b_h, b_y);
  }

  // ================= pyramid pooling + fusion =================
  poolw_kernel<<<cdiv(4 * NODES, T), T, 0, stream>>>(b_pw);
  pooled_kernel<<<cdiv((size_t)4 * B * HID, T), T, 0, stream>>>(b_h, b_pw, b_pool, B);
  proj_kernel<<<cdiv((size_t)B * 320, T), T, 0, stream>>>(
      b_pool, projW[0], projW[1], projW[2], projW[3],
      projB[0], projB[1], projB[2], projB[3], b_fused, B);
  gmeanmax_kernel<<<cdiv((size_t)B * HID, T), T, 0, stream>>>(b_h, b_fused, B);
  conv(b_fused, b_fus16, (size_t)B * 960);
  conv(fus_W, b_w16, (size_t)960 * HID);
  gemm(b_fus16, b_w16, fus_b, b_fo, nullptr, B, HID, 960, 0, 0);
  ln_kernel<<<cdiv(B, 8), 256, 0, stream>>>(b_fo, fus_g, fus_be, (float*)d_out, nullptr, B, 1);
}
